// MultiHeadAttentionLayer_83820581748993
// MI455X (gfx1250) — matrix operand order repaired and checked by reading, not compiled, not run on hardware
//
#include <hip/hip_runtime.h>
#include <stdint.h>

typedef __bf16 bf16;
typedef __bf16 v16bf __attribute__((ext_vector_type(16)));
typedef __bf16 v8bf  __attribute__((ext_vector_type(8)));
typedef __bf16 v4bf  __attribute__((ext_vector_type(4)));
typedef float  v8f   __attribute__((ext_vector_type(8)));
typedef float  v4f   __attribute__((ext_vector_type(4)));
typedef int    v4i   __attribute__((ext_vector_type(4)));

#define B_   4
#define S_   2048
#define D_   1024
#define H_   16
#define DH_  64
#define NEG_BIG (-1.0e30f)

static __device__ __forceinline__ v16bf make16(v8bf lo, v8bf hi) {
  union { v16bf v; v8bf h[2]; } u;
  u.h[0] = lo; u.h[1] = hi;
  return u.v;
}

static __device__ __forceinline__ v8f wmma_bf16(v16bf a, v16bf b, v8f c) {
  return __builtin_amdgcn_wmma_f32_16x16x32_bf16(
      /*neg_a=*/false, a, /*neg_b=*/false, b,
      /*c_mod=*/(short)0, c, /*reuse_a=*/false, /*reuse_b=*/false);
}

// --- async global->LDS copy (CDNA5 GLOBAL_LOAD_ASYNC_TO_LDS_B128) ----------
#if __has_builtin(__builtin_amdgcn_global_load_async_to_lds_b128)
#define ASYNC_COPY16(dst_lds, src_glob)                                      \
  __builtin_amdgcn_global_load_async_to_lds_b128(                            \
      (__attribute__((address_space(1))) v4i*)(src_glob),                    \
      (__attribute__((address_space(3))) v4i*)(dst_lds), 0, 0)
#define ASYNC_WAIT() asm volatile("s_wait_asynccnt 0" ::: "memory")
#else
#define ASYNC_COPY16(dst_lds, src_glob)                                      \
  (*(v8bf*)(dst_lds) = *(const v8bf*)(src_glob))
#define ASYNC_WAIT() ((void)0)
#endif

// ---------------------------------------------------------------------------
// Kernel 1: P = X @ W + bias, fp32 inputs -> bf16 output, head-split layout.
//   vmode==0:  out[b][h][s][dh]   (Q, K)
//   vmode==1:  out[b][h][dh][s]   (V transposed)
// Tile 128x128, K-step 32, 256 thr = 8 waves (4x2), 2x4 WMMA accs per wave.
// Global loads for step k+1 are issued into registers while step k computes.
// B tile is transposed at the global-load (column-strided dword gather,
// coalesced across lanes) so LDS staging is packed b64 stores.
// ---------------------------------------------------------------------------
__global__ __launch_bounds__(256) void proj_kernel(
    const float* __restrict__ X, const float* __restrict__ W,
    const float* __restrict__ bias, bf16* __restrict__ out, int vmode) {

  __shared__ bf16 As[128][40];   // rows m, cols k (pad -> 80B row stride)
  __shared__ bf16 Bt[128][40];   // rows n, cols k (transposed W tile)

  const int t     = threadIdx.x;
  const int wave  = t >> 5;
  const int lane  = t & 31;
  const int lhalf = lane & 15;
  const bool hi   = lane >= 16;
  const int wm = wave >> 1;      // 0..3 -> 32-row band
  const int wn = wave & 1;       // 0..1 -> 64-col band
  const int m0 = blockIdx.y * 128;
  const int n0 = blockIdx.x * 128;

  v8f acc[2][4];
#pragma unroll
  for (int a = 0; a < 2; ++a)
#pragma unroll
    for (int c = 0; c < 4; ++c)
      acc[a][c] = (v8f){0.f, 0.f, 0.f, 0.f, 0.f, 0.f, 0.f, 0.f};

  v4f ra[4], rb[4];
  auto load_tiles = [&](int k0) {
#pragma unroll
    for (int i = 0; i < 4; ++i) {
      int e = t + i * 256;
      // A: row-major float4 (8 granules per 32-wide row)
      int arow = e >> 3, ac4 = (e & 7) << 2;
      ra[i] = *(const v4f*)(X + (size_t)(m0 + arow) * D_ + k0 + ac4);
      // B: gather a 4-deep k-column for output column n (transposed read)
      int bn = e & 127, bk4 = (e >> 7) << 2;
      const float* wp = W + (size_t)(k0 + bk4) * D_ + n0 + bn;
      rb[i] = (v4f){ wp[0], wp[D_], wp[2 * D_], wp[3 * D_] };
    }
  };

  load_tiles(0);

  for (int k0 = 0; k0 < D_; k0 += 32) {
    __syncthreads();                 // prior iteration's fragment reads done
    // convert staged registers -> LDS (bf16)
#pragma unroll
    for (int i = 0; i < 4; ++i) {
      int e = t + i * 256;
      int arow = e >> 3, ac4 = (e & 7) << 2;
      v4bf ha = { (bf16)ra[i][0], (bf16)ra[i][1],
                  (bf16)ra[i][2], (bf16)ra[i][3] };
      *(v4bf*)&As[arow][ac4] = ha;
      int bn = e & 127, bk4 = (e >> 7) << 2;
      v4bf hb = { (bf16)rb[i][0], (bf16)rb[i][1],
                  (bf16)rb[i][2], (bf16)rb[i][3] };
      *(v4bf*)&Bt[bn][bk4] = hb;
    }
    __syncthreads();
    if (k0 + 32 < D_) load_tiles(k0 + 32);   // overlap with WMMA below

    // A fragments (2 m-subtiles)
    v16bf af[2];
#pragma unroll
    for (int mf = 0; mf < 2; ++mf) {
      int row = wm * 32 + mf * 16 + lhalf;
      int kb  = hi ? 8 : 0;
      af[mf] = make16(*(const v8bf*)&As[row][kb],
                      *(const v8bf*)&As[row][16 + kb]);
    }
    // B fragments (4 n-subtiles) + MMA
#pragma unroll
    for (int nf = 0; nf < 4; ++nf) {
      int col = wn * 64 + nf * 16 + lhalf;
      int kb  = hi ? 8 : 0;
      v16bf bfr = make16(*(const v8bf*)&Bt[col][kb],
                         *(const v8bf*)&Bt[col][kb + 16]);
#pragma unroll
      for (int mf = 0; mf < 2; ++mf)
        acc[mf][nf] = wmma_bf16(af[mf], bfr, acc[mf][nf]);
    }
  }

  // Epilogue: + bias, bf16, scatter to head-split layout.
  const int r0 = hi ? 8 : 0;
#pragma unroll
  for (int nf = 0; nf < 4; ++nf) {
    int col  = n0 + wn * 64 + nf * 16 + lhalf;   // global feature index
    float bv = bias[col];
    int head = col >> 6;
    int dh   = col & 63;
#pragma unroll
    for (int mf = 0; mf < 2; ++mf) {
#pragma unroll
      for (int i = 0; i < 8; ++i) {
        int grow = m0 + wm * 32 + mf * 16 + r0 + i;  // global row (b*S+s)
        int b = grow >> 11;
        int s = grow & (S_ - 1);
        float val = acc[mf][nf][i] + bv;
        size_t addr = vmode
            ? ((((size_t)b * H_ + head) * DH_ + dh) * S_ + s)   // V^T
            : ((((size_t)b * H_ + head) * S_ + s) * DH_ + dh);  // Q, K
        out[addr] = (bf16)val;
      }
    }
  }
}

// ---------------------------------------------------------------------------
// Kernel 2: flash attention. grid = (N/64, H, B), 128 thr = 4 waves.
// K/V tiles double-buffered in LDS via GLOBAL_LOAD_ASYNC_TO_LDS_B128:
// block i+1's copies are in flight while block i computes (ASYNCcnt split
// issue/wait). Each wave: 16 query rows, full Dh=64, online softmax.
// ---------------------------------------------------------------------------
__device__ __forceinline__ void reduce_max16(float r[8]) {
#pragma unroll
  for (int s = 1; s < 16; s <<= 1)
#pragma unroll
    for (int i = 0; i < 8; ++i)
      r[i] = fmaxf(r[i], __shfl_xor(r[i], s, 32));
}
__device__ __forceinline__ void reduce_sum16(float r[8]) {
#pragma unroll
  for (int s = 1; s < 16; s <<= 1)
#pragma unroll
    for (int i = 0; i < 8; ++i)
      r[i] += __shfl_xor(r[i], s, 32);
}

__global__ __launch_bounds__(128) void attn_kernel(
    const bf16* __restrict__ Qp, const bf16* __restrict__ Kp,
    const bf16* __restrict__ Vt, const int* __restrict__ mask,
    float* __restrict__ out) {

  __shared__ bf16 Ks[2][64][72];   // [buf][key_local][feat]   (pad 16B)
  __shared__ bf16 Vs[2][64][72];   // [buf][feat][key_local]
  __shared__ bf16 Ps[4][16][72];   // per-wave P tile

  const int wave  = threadIdx.x >> 5;
  const int lane  = threadIdx.x & 31;
  const int lhalf = lane & 15;
  const bool hi   = lane >= 16;
  const int r0    = hi ? 8 : 0;
  const int b = blockIdx.z, h = blockIdx.y;
  const int qbase = blockIdx.x * 64 + wave * 16;

  const bf16* Q = Qp + ((size_t)b * H_ + h) * S_ * DH_;
  const bf16* K = Kp + ((size_t)b * H_ + h) * S_ * DH_;
  const bf16* V = Vt + ((size_t)b * H_ + h) * DH_ * S_;

  auto stage = [&](int buf, int kb0) {
#pragma unroll
    for (int i = 0; i < 4; ++i) {
      int c   = threadIdx.x + i * 128;  // chunk 0..511
      int row = c >> 3;                 // 8 chunks per 64-elem row
      int col = (c & 7) << 3;
      ASYNC_COPY16(&Ks[buf][row][col], K + (size_t)(kb0 + row) * DH_ + col);
      ASYNC_COPY16(&Vs[buf][row][col], V + (size_t)row * S_ + kb0 + col);
    }
  };

  // Q A-fragments, feature steps ks=0 (0..31), ks=1 (32..63).
  v16bf qf[2];
  {
    const bf16* qr = Q + (size_t)(qbase + lhalf) * DH_;
#pragma unroll
    for (int ks = 0; ks < 2; ++ks) {
      int kb = ks * 32 + (hi ? 8 : 0);
      qf[ks] = make16(*(const v8bf*)(qr + kb), *(const v8bf*)(qr + kb + 16));
    }
  }

  v8f o[4];
#pragma unroll
  for (int nt = 0; nt < 4; ++nt)
    o[nt] = (v8f){0.f, 0.f, 0.f, 0.f, 0.f, 0.f, 0.f, 0.f};
  float m_i[8], l_i[8];
#pragma unroll
  for (int i = 0; i < 8; ++i) { m_i[i] = NEG_BIG; l_i[i] = 0.f; }

  stage(0, 0);
  ASYNC_WAIT();
  __syncthreads();

  for (int kb0 = 0; kb0 < S_; kb0 += 64) {
    const int cur = (kb0 >> 6) & 1;
    if (kb0 + 64 < S_) stage(cur ^ 1, kb0 + 64);   // prefetch next block

    // ---- scores S = (Q K^T)/8 for 16 queries x 64 keys -------------------
    v8f sc[4];
#pragma unroll
    for (int jt = 0; jt < 4; ++jt) {
      sc[jt] = (v8f){0.f, 0.f, 0.f, 0.f, 0.f, 0.f, 0.f, 0.f};
      const bf16* kr = &Ks[cur][jt * 16 + lhalf][0];
#pragma unroll
      for (int ks = 0; ks < 2; ++ks) {
        int fb = ks * 32 + (hi ? 8 : 0);
        v16bf bfr = make16(*(const v8bf*)(kr + fb), *(const v8bf*)(kr + fb + 16));
        sc[jt] = wmma_bf16(qf[ks], bfr, sc[jt]);
      }
    }
    // scale + mask (column == key per lane)
#pragma unroll
    for (int jt = 0; jt < 4; ++jt) {
      int key = kb0 + jt * 16 + lhalf;
      bool keep = mask[key] != 0;
#pragma unroll
      for (int i = 0; i < 8; ++i)
        sc[jt][i] = keep ? sc[jt][i] * 0.125f : NEG_BIG;
    }
    // ---- online softmax --------------------------------------------------
    float bm[8];
#pragma unroll
    for (int i = 0; i < 8; ++i)
      bm[i] = fmaxf(fmaxf(sc[0][i], sc[1][i]), fmaxf(sc[2][i], sc[3][i]));
    reduce_max16(bm);
    float alpha[8], rs[8];
#pragma unroll
    for (int i = 0; i < 8; ++i) {
      float mn = fmaxf(m_i[i], bm[i]);
      alpha[i] = __expf(m_i[i] - mn);
      m_i[i] = mn;
      rs[i] = 0.f;
    }
#pragma unroll
    for (int jt = 0; jt < 4; ++jt)
#pragma unroll
      for (int i = 0; i < 8; ++i) {
        float p = __expf(sc[jt][i] - m_i[i]);
        sc[jt][i] = p;
        rs[i] += p;
      }
    reduce_sum16(rs);
#pragma unroll
    for (int i = 0; i < 8; ++i) l_i[i] = l_i[i] * alpha[i] + rs[i];
#pragma unroll
    for (int nt = 0; nt < 4; ++nt)
#pragma unroll
      for (int i = 0; i < 8; ++i) o[nt][i] *= alpha[i];

    // ---- P -> LDS (C layout -> row major), reload as A fragments ---------
#pragma unroll
    for (int jt = 0; jt < 4; ++jt)
#pragma unroll
      for (int i = 0; i < 8; ++i)
        Ps[wave][r0 + i][jt * 16 + lhalf] = (bf16)sc[jt][i];
    asm volatile("s_wait_dscnt 0" ::: "memory");

    v16bf pf[2];
#pragma unroll
    for (int ks = 0; ks < 2; ++ks) {
      const bf16* pp = &Ps[wave][lhalf][ks * 32 + (hi ? 8 : 0)];
      pf[ks] = make16(*(const v8bf*)pp, *(const v8bf*)(pp + 16));
    }

    // ---- O += P @ V ------------------------------------------------------
#pragma unroll
    for (int nt = 0; nt < 4; ++nt) {
      const bf16* vr = &Vs[cur][nt * 16 + lhalf][0];
#pragma unroll
      for (int ks = 0; ks < 2; ++ks) {
        int kk = ks * 32 + (hi ? 8 : 0);
        v16bf bfr = make16(*(const v8bf*)(vr + kk), *(const v8bf*)(vr + kk + 16));
        o[nt] = wmma_bf16(pf[ks], bfr, o[nt]);
      }
    }

    ASYNC_WAIT();       // prefetched K/V for next block landed in LDS
    __syncthreads();    // all waves done reading cur before it is reused
  }

  // ---- finalize: O / l, fp32 to [B, N, H*Dh] -------------------------------
  float inv[8];
#pragma unroll
  for (int i = 0; i < 8; ++i) inv[i] = 1.0f / l_i[i];
#pragma unroll
  for (int nt = 0; nt < 4; ++nt) {
    int feat = h * DH_ + nt * 16 + lhalf;
#pragma unroll
    for (int i = 0; i < 8; ++i) {
      int row = qbase + r0 + i;
      out[((size_t)b * S_ + row) * (H_ * DH_) + feat] = o[nt][i] * inv[i];
    }
  }
}

// ---------------------------------------------------------------------------
extern "C" void kernel_launch(void* const* d_in, const int* in_sizes, int n_in,
                              void* d_out, int out_size, void* d_ws, size_t ws_size,
                              hipStream_t stream) {
  (void)in_sizes; (void)n_in; (void)out_size; (void)ws_size;
  const float* q  = (const float*)d_in[0];
  const float* k  = (const float*)d_in[1];
  const float* v  = (const float*)d_in[2];
  const float* Wq = (const float*)d_in[3];
  const float* bq = (const float*)d_in[4];
  const float* Wk = (const float*)d_in[5];
  const float* bk = (const float*)d_in[6];
  const float* Wv = (const float*)d_in[7];
  const float* bv = (const float*)d_in[8];
  const int* mask = (const int*)d_in[9];
  float* out = (float*)d_out;

  const size_t elems = (size_t)B_ * H_ * S_ * DH_;   // 8,388,608 bf16 each
  bf16* Qp = (bf16*)d_ws;
  bf16* Kp = Qp + elems;
  bf16* Vt = Kp + elems;                              // 48 MB total workspace

  dim3 pgrid(D_ / 128, (B_ * S_) / 128);              // (8, 64)
  proj_kernel<<<pgrid, 256, 0, stream>>>(q, Wq, bq, Qp, 0);
  proj_kernel<<<pgrid, 256, 0, stream>>>(k, Wk, bk, Kp, 0);
  proj_kernel<<<pgrid, 256, 0, stream>>>(v, Wv, bv, Vt, 1);

  dim3 agrid(S_ / 64, H_, B_);                        // (32, 16, 4)
  attn_kernel<<<agrid, 128, 0, stream>>>(Qp, Kp, Vt, mask, out);
}
